// WGCN_53919019434432
// MI455X (gfx1250) — matrix operand order repaired and checked by reading, not compiled, not run on hardware
//
#include <hip/hip_runtime.h>
#include <hip/hip_bf16.h>
#include <math.h>

// ---------------- CDNA5 WMMA types ----------------
typedef __attribute__((ext_vector_type(16))) __bf16 v16bf;
typedef __attribute__((ext_vector_type(8)))  __bf16 v8bf;
typedef __attribute__((ext_vector_type(4)))  __bf16 v4bf;
typedef __attribute__((ext_vector_type(8)))  float  v8f;

#define F_IN  500
#define NKT1  16            // ceil(500/32) K-tiles for layer 1
#define HID   64
#define NCLS  40

// ---------------- degree / normalization ----------------
__global__ void k_init_deg(float* __restrict__ deg, int n) {
    int i = blockIdx.x * blockDim.x + threadIdx.x;
    if (i < n) deg[i] = 1.0f;                       // self-loop contributes 1
}

__global__ void k_deg_edges(const int* __restrict__ ei, float* __restrict__ deg,
                            int E) {
    int e = blockIdx.x * blockDim.x + threadIdx.x;
    if (e < E) atomicAdd(&deg[ei[E + e]], 1.0f);    // dst row of edge_index
}

__global__ void k_dinv(const float* __restrict__ deg, float* __restrict__ dinv,
                       int n) {
    int i = blockIdx.x * blockDim.x + threadIdx.x;
    if (i < n) dinv[i] = rsqrtf(deg[i]);            // deg >= 1 always
}

// ------- weight pre-pack: fp32 -> bf16 in exact WMMA B-fragment order -------
// Layout: [kt][wave][lane][e] ; fragment element e of lane L holds
// B(K = kt*32 + (L>>4)*8 + (e&7) + (e>>3)*16, N = wave*16 + (L&15)). One lane fragment =
// 16 contiguous bf16 = 32B -> loads as 2x global_load_b128.
__global__ void k_w1_pack(const float* __restrict__ W1, __bf16* __restrict__ P) {
    int t = blockIdx.x * blockDim.x + threadIdx.x;  // NKT1*4*32*16 = 32768
    if (t >= NKT1 * 4 * 32 * 16) return;
    int e    = t & 15;
    int lane = (t >> 4) & 31;
    int w    = (t >> 9) & 3;
    int kt   = t >> 11;
    int K = kt * 32 + ((lane >> 4) << 3) + (e & 7) + ((e >> 3) << 4);
    int N = (w << 4) + (lane & 15);
    float v = (K < F_IN) ? W1[K * HID + N] : 0.0f;
    P[t] = (__bf16)v;
}

__global__ void k_w2_pack(const float* __restrict__ W2, __bf16* __restrict__ P) {
    int t = blockIdx.x * blockDim.x + threadIdx.x;  // 2*3*32*16 = 3072
    if (t >= 2 * 3 * 32 * 16) return;
    int e    = t & 15;
    int lane = (t >> 4) & 31;
    int wk   = t >> 9;                              // 0..5
    int kt   = wk / 3;
    int w    = wk - kt * 3;
    int K = kt * 32 + ((lane >> 4) << 3) + (e & 7) + ((e >> 3) << 4);       // < 64 always
    int N = (w << 4) + (lane & 15);
    float v = (N < NCLS) ? W2[K * NCLS + N] : 0.0f;
    P[t] = (__bf16)v;
}

// ---------------- GEMM1: h1 = x @ W1  (bf16 WMMA, fp32 accum) ----------------
// grid.x = ceil(N/16); 4 waves/block, wave w owns N-columns [16w,16w+16).
// A tile staged in LDS as bf16, already permuted into fragment order:
//   pos(K) = ((K>>3)&1)*16 + ((K>>4)&1)*8 + (K&7)
// so lane (g,ln) reads its v16bf fragment as two ds_load_b128 at
// As[ln][g*16]. Pitch 40 bf16 (80B): lanes 0-15 tile all 64 LDS banks.
__global__ __launch_bounds__(128) void k_gemm1(const float*  __restrict__ X,
                                               const v16bf* __restrict__ Bf,
                                               float* __restrict__ H, int n) {
    const int m0   = blockIdx.x << 4;
    const int lane = threadIdx.x & 31;
    const int wave = threadIdx.x >> 5;
    const int n0   = wave << 4;
    const int g    = lane >> 4;
    const int ln   = lane & 15;
    __shared__ __bf16 As[16][40];

    // cooperative loader mapping: 128 threads x float4 = 16x32 tile
    const int t   = threadIdx.x;
    const int row = t >> 3;
    const int kq  = (t & 7) << 2;                   // K quad within 32-tile
    const int pos = (((kq >> 3) & 1) << 4) + (((kq >> 4) & 1) << 3) + (kq & 7);
    const int m   = m0 + row;
    const float* xrow = X + (size_t)(m < n ? m : 0) * F_IN;
    const int kmax = (m < n) ? F_IN : 0;            // zero rows beyond M

    v8f acc = {};
    for (int kt = 0; kt < NKT1; ++kt) {
        const int k0 = kt << 5;
        // F_IN % 4 == 0 -> a float4 is fully in range or fully out
        float4 v = make_float4(0.f, 0.f, 0.f, 0.f);
        if (k0 + kq < kmax) v = *(const float4*)(xrow + k0 + kq);
        v4bf pk;
        pk[0] = (__bf16)v.x; pk[1] = (__bf16)v.y;
        pk[2] = (__bf16)v.z; pk[3] = (__bf16)v.w;
        *(v4bf*)&As[row][pos] = pk;                 // 8B ds_store
        __syncthreads();

        const v8bf* ap = (const v8bf*)&As[ln][g << 4];
        v8bf alo = ap[0];                           // ds_load_b128
        v8bf ahi = ap[1];                           // ds_load_b128
        v16bf a = __builtin_shufflevector(alo, ahi, 0, 1, 2, 3, 4, 5, 6, 7,
                                          8, 9, 10, 11, 12, 13, 14, 15);
        v16bf b = Bf[(kt * 4 + wave) * 32 + lane];  // 2x global_load_b128
        acc = __builtin_amdgcn_wmma_f32_16x16x32_bf16(false, a, false, b,
                                                      (short)0, acc, false, false);
        __syncthreads();
    }
    // C/D layout: VGPR r -> M = r + 8*g, N = ln
#pragma unroll
    for (int r = 0; r < 8; ++r) {
        int mm = m0 + (g << 3) + r;
        if (mm < n) H[(size_t)mm * HID + n0 + ln] = acc[r];
    }
}

// ---------------- aggregation layer 1 ----------------
__global__ void k_agg1_init(const float* __restrict__ h1,
                            const float* __restrict__ dinv,
                            const float* __restrict__ b1,
                            float* __restrict__ agg, int n) {
    int t = blockIdx.x * blockDim.x + threadIdx.x;
    if (t >= n * HID) return;
    int i = t >> 6, j = t & 63;
    float di = dinv[i];
    agg[t] = b1[j] + di * di * h1[t];               // self-loop + bias
}

__global__ void k_scatter1(const int* __restrict__ ei,
                           const float* __restrict__ h1,
                           const float* __restrict__ dinv,
                           float* __restrict__ agg, int E) {
    int t = blockIdx.x * blockDim.x + threadIdx.x;  // 16 threads per edge
    if (t >= E * 16) return;
    int e = t >> 4, q = t & 15;
    int s = ei[e], d = ei[E + e];
    float nrm = dinv[s] * dinv[d];
    const float4 v = *(const float4*)(h1 + (size_t)s * HID + (q << 2));
    float* o = agg + (size_t)d * HID + (q << 2);
    atomicAdd(o + 0, nrm * v.x);
    atomicAdd(o + 1, nrm * v.y);
    atomicAdd(o + 2, nrm * v.z);
    atomicAdd(o + 3, nrm * v.w);
}

__global__ void k_relu_bf(const float* __restrict__ agg,
                          __bf16* __restrict__ out, int cnt) {
    int t = blockIdx.x * blockDim.x + threadIdx.x;
    if (t < cnt) out[t] = (__bf16)fmaxf(agg[t], 0.0f);
}

// ---------------- GEMM2: h2 = relu(agg1) @ W2  (bf16 WMMA) ----------------
// grid.x = ceil(N/16); 3 waves/block cover 48 padded columns (store N<40).
// A is row-major bf16 [n][64]; a lane's fragment is two contiguous 16B
// chunks (K = k0+8g..+7 and k0+16+8g..+7) -> 2x global_load_b128.
__global__ __launch_bounds__(96) void k_gemm2(const __bf16* __restrict__ A,
                                              const v16bf* __restrict__ Bf,
                                              float* __restrict__ H, int n) {
    const int m0   = blockIdx.x << 4;
    const int lane = threadIdx.x & 31;
    const int wave = threadIdx.x >> 5;
    const int n0   = wave << 4;
    const int g    = lane >> 4;
    const int ln   = lane & 15;
    const int m    = m0 + ln;
    const __bf16* arow = A + (size_t)(m < n ? m : 0) * HID;  // clamped; stores guarded

    v8f acc = {};
#pragma unroll
    for (int kt = 0; kt < 2; ++kt) {
        const int k0 = kt << 5;
        v8bf alo = *(const v8bf*)(arow + k0 + (g << 3));
        v8bf ahi = *(const v8bf*)(arow + k0 + 16 + (g << 3));
        v16bf a = __builtin_shufflevector(alo, ahi, 0, 1, 2, 3, 4, 5, 6, 7,
                                          8, 9, 10, 11, 12, 13, 14, 15);
        v16bf b = Bf[(kt * 3 + wave) * 32 + lane];
        acc = __builtin_amdgcn_wmma_f32_16x16x32_bf16(false, a, false, b,
                                                      (short)0, acc, false, false);
    }
    int nn = n0 + ln;
    if (nn < NCLS) {
#pragma unroll
        for (int r = 0; r < 8; ++r) {
            int mm = m0 + (g << 3) + r;
            if (mm < n) H[(size_t)mm * NCLS + nn] = acc[r];
        }
    }
}

// ---------------- aggregation layer 2 ----------------
__global__ void k_agg2_init(const float* __restrict__ h2,
                            const float* __restrict__ dinv,
                            const float* __restrict__ b2,
                            float* __restrict__ agg, int n) {
    int t = blockIdx.x * blockDim.x + threadIdx.x;
    if (t >= n * NCLS) return;
    int i = t / NCLS, c = t - i * NCLS;
    float di = dinv[i];
    agg[t] = b2[c] + di * di * h2[t];
}

__global__ void k_scatter2(const int* __restrict__ ei,
                           const float* __restrict__ h2,
                           const float* __restrict__ dinv,
                           float* __restrict__ agg, int E) {
    int t = blockIdx.x * blockDim.x + threadIdx.x;  // 16 slots/edge, 10 active
    if (t >= E * 16) return;
    int e = t >> 4, q = t & 15;
    if (q >= 10) return;                            // 10 * float4 = 40 classes
    int s = ei[e], d = ei[E + e];
    float nrm = dinv[s] * dinv[d];
    const float4 v = *(const float4*)(h2 + (size_t)s * NCLS + (q << 2));
    float* o = agg + (size_t)d * NCLS + (q << 2);
    atomicAdd(o + 0, nrm * v.x);
    atomicAdd(o + 1, nrm * v.y);
    atomicAdd(o + 2, nrm * v.z);
    atomicAdd(o + 3, nrm * v.w);
}

// ---------------- row-wise log_softmax over 40 classes ----------------
__global__ void k_logsoftmax(const float* __restrict__ agg,
                             float* __restrict__ out, int n) {
    int i = blockIdx.x * blockDim.x + threadIdx.x;
    if (i >= n) return;
    const float* r = agg + (size_t)i * NCLS;
    float mx = -INFINITY;
#pragma unroll
    for (int c = 0; c < NCLS; ++c) mx = fmaxf(mx, r[c]);
    float s = 0.0f;
#pragma unroll
    for (int c = 0; c < NCLS; ++c) s += __expf(r[c] - mx);
    float lse = mx + __logf(s);
    float* o = out + (size_t)i * NCLS;
#pragma unroll
    for (int c = 0; c < NCLS; ++c) o[c] = r[c] - lse;
}

// ---------------- host launcher ----------------
extern "C" void kernel_launch(void* const* d_in, const int* in_sizes, int n_in,
                              void* d_out, int out_size, void* d_ws, size_t ws_size,
                              hipStream_t stream) {
    const float* x  = (const float*)d_in[0];
    const float* W1 = (const float*)d_in[1];
    const float* b1 = (const float*)d_in[2];
    const float* W2 = (const float*)d_in[3];
    const float* b2 = (const float*)d_in[4];
    const int*   ei = (const int*)d_in[5];

    const int N = in_sizes[0] / F_IN;    // 100000
    const int E = in_sizes[5] / 2;       // 1600000

    // workspace carve-out (256B aligned regions)
    char* w = (char*)d_ws;
    auto carve = [&](size_t bytes) {
        void* p = (void*)w;
        w += (bytes + 255) & ~(size_t)255;
        return p;
    };
    float*  h1   = (float*) carve((size_t)N * HID * 4);
    float*  agg1 = (float*) carve((size_t)N * HID * 4);
    __bf16* hbf  = (__bf16*)carve((size_t)N * HID * 2);
    float*  h2   = (float*) carve((size_t)N * NCLS * 4);
    float*  o2   = (float*) carve((size_t)N * NCLS * 4);
    float*  deg  = (float*) carve((size_t)N * 4);
    float*  dinv = (float*) carve((size_t)N * 4);
    __bf16* W1p  = (__bf16*)carve((size_t)NKT1 * 4 * 32 * 16 * 2);
    __bf16* W2p  = (__bf16*)carve((size_t)2 * 3 * 32 * 16 * 2);

    const int TB = 256;
    // 1) degrees + normalization
    k_init_deg <<<(N + TB - 1) / TB, TB, 0, stream>>>(deg, N);
    k_deg_edges<<<(E + TB - 1) / TB, TB, 0, stream>>>(ei, deg, E);
    k_dinv     <<<(N + TB - 1) / TB, TB, 0, stream>>>(deg, dinv, N);
    // 2) weight fragment pre-pack
    k_w1_pack  <<<(NKT1 * 4 * 32 * 16 + TB - 1) / TB, TB, 0, stream>>>(W1, W1p);
    k_w2_pack  <<<(2 * 3 * 32 * 16 + TB - 1) / TB, TB, 0, stream>>>(W2, W2p);
    // 3) layer 1
    k_gemm1    <<<(N + 15) / 16, 128, 0, stream>>>(x, (const v16bf*)W1p, h1, N);
    k_agg1_init<<<((size_t)N * HID + TB - 1) / TB, TB, 0, stream>>>(h1, dinv, b1, agg1, N);
    k_scatter1 <<<((size_t)E * 16 + TB - 1) / TB, TB, 0, stream>>>(ei, h1, dinv, agg1, E);
    k_relu_bf  <<<((size_t)N * HID + TB - 1) / TB, TB, 0, stream>>>(agg1, hbf, N * HID);
    // 4) layer 2
    k_gemm2    <<<(N + 15) / 16, 96, 0, stream>>>(hbf, (const v16bf*)W2p, h2, N);
    k_agg2_init<<<((size_t)N * NCLS + TB - 1) / TB, TB, 0, stream>>>(h2, dinv, b2, o2, N);
    k_scatter2 <<<((size_t)E * 16 + TB - 1) / TB, TB, 0, stream>>>(ei, h2, dinv, o2, E);
    // 5) log_softmax
    k_logsoftmax<<<(N + TB - 1) / TB, TB, 0, stream>>>(o2, (float*)d_out, N);
}
